// SpatialGraphConvolution_79053168050685
// MI455X (gfx1250) — matrix operand order repaired and checked by reading, not compiled, not run on hardware
//
#include <hip/hip_runtime.h>
#include <hip/hip_bf16.h>

typedef __attribute__((ext_vector_type(16))) __bf16 v16bf;
typedef __attribute__((ext_vector_type(8)))  float  v8f;
typedef __attribute__((ext_vector_type(4)))  unsigned int u32x4;
typedef __attribute__((ext_vector_type(8)))  unsigned int u32x8;

__device__ __forceinline__ v16bf frag_2x8(const __bf16* p0, const __bf16* p1) {
    union { v16bf v; uint4 q[2]; } f;
    f.q[0] = *(const uint4*)p0;   // 16B contiguous
    f.q[1] = *(const uint4*)p1;   // 16B contiguous
    return f.v;
}

__device__ __forceinline__ v8f wmma_bf16(v16bf a, v16bf b, v8f c) {
    return __builtin_amdgcn_wmma_f32_16x16x32_bf16(false, a, false, b, (short)0, c, false, false);
}

// TDM: copy one contiguous 2KB tile (1024 x bf16) global -> LDS via the
// Tensor Data Mover (D# per CDNA5 ISA ch.8; type=2, data_size=2B, 1-row tile).
__device__ __forceinline__ void tdm_load_2kb(const void* gsrc, void* lds_dst) {
    unsigned long long ga = (unsigned long long)gsrc;
    unsigned int lds = (unsigned int)(unsigned long long)lds_dst;   // LDS byte offset
    u32x4 g0;
    g0[0] = 1u;                                                     // count=1 (valid user D#)
    g0[1] = lds;                                                    // lds_addr
    g0[2] = (unsigned int)ga;                                       // global_addr[31:0]
    g0[3] = ((unsigned int)(ga >> 32) & 0x01FFFFFFu) | 0x80000000u; // addr[56:32] | type=2
    u32x8 g1;
    g1[0] = 0x00010000u;        // workgroup_mask=0, data_size=1 (2 bytes)
    g1[1] = (1024u << 16);      // tensor_dim0 = 1024 (low 16)
    g1[2] = (1u    << 16);      // tensor_dim0 hi=0, tensor_dim1 = 1
    g1[3] = (1024u << 16);      // tensor_dim1 hi=0, tile_dim0 = 1024
    g1[4] = 1u;                 // tile_dim1 = 1, tile_dim2 = 0
    g1[5] = 1024u;              // tensor_dim0_stride = 1024
    g1[6] = 0u;
    g1[7] = 0u;
    u32x4 gz = {0u, 0u, 0u, 0u};
    asm volatile("tensor_load_to_lds %0, %1, %2, %3"
                 :: "s"(g0), "s"(g1), "s"(gz), "s"(gz)
                 : "memory");
}

// ---------------------------------------------------------------------------
// Kernel 0: convert small weight tensors to bf16 in workspace
// ---------------------------------------------------------------------------
__global__ void k_convert_weights(const float* __restrict__ wth,
                                  const float* __restrict__ wph,
                                  const float* __restrict__ wo,
                                  __bf16* __restrict__ wth_bf,
                                  __bf16* __restrict__ wph_bf,
                                  __bf16* __restrict__ wo_bf) {
    int i = blockIdx.x * 256 + threadIdx.x;
    if (i < 6144)        wth_bf[i]         = (__bf16)wth[i];
    else if (i < 12288)  wph_bf[i - 6144]  = (__bf16)wph[i - 6144];
    else if (i < 36864)  wo_bf[i - 12288]  = (__bf16)wo[i - 12288];
}

// ---------------------------------------------------------------------------
// Kernel 1: per (k,b): c = softmax_v(theta^T phi / 9600), adj = A+Bm+c,
// written transposed & v-padded as bf16 [u][32].  256 threads = 8 waves.
// ---------------------------------------------------------------------------
__global__ __launch_bounds__(256) void k_attention(
    const float* __restrict__ x,        // (64,64,300,25)
    const float* __restrict__ A,        // (3,25,25)
    const float* __restrict__ Bm,       // (3,25,25)
    const __bf16* __restrict__ wth_bf,  // (3,32,64)
    const __bf16* __restrict__ wph_bf,  // (3,32,64)
    const float* __restrict__ b_theta,  // (3,32)
    const float* __restrict__ b_phi,    // (3,32)
    __bf16* __restrict__ adjbf)         // (3,64,32,32)
{
    __shared__ __bf16 xs[208 * 72];   // [n=(t*25+v) pad208][c]  stride 144B (conflict-free)
    __shared__ __bf16 th[32 * 264];   // [v pad32][t*32+i]       stride 528B (conflict-free)
    __shared__ __bf16 ph[32 * 264];
    __shared__ float  biasl[64];      // [which][i]
    float* cl = reinterpret_cast<float*>(xs);   // 32x32 f32, aliases dead xs at the end

    const int kb = blockIdx.x;
    const int k = kb / 64, b = kb % 64;
    const int tid  = threadIdx.x;
    const int wave = tid >> 5, lane = tid & 31;
    const int l15  = lane & 15, hh = lane >> 4;
    const v8f vzero = {0.f, 0.f, 0.f, 0.f, 0.f, 0.f, 0.f, 0.f};

    // one-time pad zeroing + bias preload
    for (int i = tid; i < 8 * 72; i += 256) xs[200 * 72 + i] = (__bf16)0.f;
    for (int i = tid; i < 7 * 264; i += 256) { th[25 * 264 + i] = (__bf16)0.f; ph[25 * 264 + i] = (__bf16)0.f; }
    if (tid < 64) biasl[tid] = (tid < 32) ? b_theta[k * 32 + tid] : b_phi[k * 32 + tid - 32];

    // weight A-fragments: invariant for whole block -> keep in registers
    v16bf wfrag[2][2][2];   // [which][mt][kk]
#pragma unroll
    for (int which = 0; which < 2; ++which) {
        const __bf16* W = which ? wph_bf : wth_bf;
#pragma unroll
        for (int mt = 0; mt < 2; ++mt) {
            const __bf16* arow = W + (k * 32 + mt * 16 + l15) * 64;
#pragma unroll
            for (int kk = 0; kk < 2; ++kk)
                wfrag[which][mt][kk] =
                    frag_2x8(arow + kk * 32 + 8 * hh, arow + kk * 32 + 16 + 8 * hh);
        }
    }

    v8f cacc = vzero;
    const int cmt = wave >> 1, cnt = wave & 1;   // c-tile ownership (waves 0..3)

    for (int chnk = 0; chnk < 38; ++chnk) {
        const int t0 = chnk * 8;
        // stage x chunk: [n=(t,v)][c] bf16 (coalesced 200-float runs per channel)
        for (int idx = tid; idx < 200 * 64; idx += 256) {
            int c = idx / 200, j = idx % 200;
            float val = (t0 * 25 + j < 7500) ? x[(b * 64 + c) * 7500 + t0 * 25 + j] : 0.f;
            xs[j * 72 + c] = (__bf16)val;
        }
        // hint-prefetch next chunk (global_prefetch_b8)
        if (t0 + 8 < 300)
            __builtin_prefetch(&x[(b * 64 + (tid & 63)) * 7500 + (t0 + 8) * 25 + (tid >> 6) * 50], 0, 0);
        __syncthreads();

        // theta/phi GEMM: 26 jobs = (nt 0..12) x (theta|phi); 4 WMMAs per job
        for (int j26 = wave; j26 < 26; j26 += 8) {
            const int which = j26 & 1;
            const int nt = j26 >> 1;
            const __bf16* brow = xs + (nt * 16 + l15) * 72;
            v16bf bf0 = frag_2x8(brow + 8 * hh, brow + 16 + 8 * hh);
            v16bf bf1 = frag_2x8(brow + 32 + 8 * hh, brow + 32 + 16 + 8 * hh);
            const int ncol = nt * 16 + l15;
            const int tl = ncol / 25, v = ncol % 25;
            const bool nvalid = ncol < 200;
            const bool tvalid = nvalid && (t0 + tl) < 300;
            __bf16* dst = which ? ph : th;
#pragma unroll
            for (int mt = 0; mt < 2; ++mt) {
                v8f acc = wmma_bf16(wfrag[which][mt][0], bf0, vzero);
                acc = wmma_bf16(wfrag[which][mt][1], bf1, acc);
                if (nvalid) {
                    const int i0 = mt * 16 + 8 * hh;
                    union { uint4 q; __bf16 e[8]; } pk;
                    if (tvalid) {
                        const float4* bp = (const float4*)&biasl[which * 32 + i0];
                        float4 b0 = bp[0], b1 = bp[1];
                        pk.e[0] = (__bf16)(acc[0] + b0.x); pk.e[1] = (__bf16)(acc[1] + b0.y);
                        pk.e[2] = (__bf16)(acc[2] + b0.z); pk.e[3] = (__bf16)(acc[3] + b0.w);
                        pk.e[4] = (__bf16)(acc[4] + b1.x); pk.e[5] = (__bf16)(acc[5] + b1.y);
                        pk.e[6] = (__bf16)(acc[6] + b1.z); pk.e[7] = (__bf16)(acc[7] + b1.w);
                    } else {
                        pk.q = (uint4){0u, 0u, 0u, 0u};
                    }
                    *(uint4*)&dst[v * 264 + tl * 32 + i0] = pk.q;   // 16B packed store
                }
            }
        }
        __syncthreads();

        // c GEMM: c[v][u] += theta^T * phi over K = t*32+i = 256 (8 k-steps)
        if (wave < 4) {
#pragma unroll
            for (int kk = 0; kk < 8; ++kk) {
                const __bf16* arow = th + (cmt * 16 + l15) * 264 + kk * 32;
                const __bf16* brow = ph + (cnt * 16 + l15) * 264 + kk * 32;
                v16bf af = frag_2x8(arow + 8 * hh, arow + 16 + 8 * hh);
                v16bf bf = frag_2x8(brow + 8 * hh, brow + 16 + 8 * hh);
                cacc = wmma_bf16(af, bf, cacc);
            }
        }
        __syncthreads();
    }

    if (wave < 4) {
#pragma unroll
        for (int r = 0; r < 8; ++r)
            cl[(cmt * 16 + r + 8 * hh) * 32 + cnt * 16 + l15] = cacc[r];
    }
    __syncthreads();

    // softmax over v per column u, add A + Bm, store transposed bf16 [u][v pad32]
    if (wave == 0) {
        const int u = lane;
        if (u < 25) {
            float col[25];
            float mx = -1e30f;
#pragma unroll
            for (int v = 0; v < 25; ++v) { col[v] = cl[v * 32 + u] * (1.f / 9600.f); mx = fmaxf(mx, col[v]); }
            float s = 0.f;
#pragma unroll
            for (int v = 0; v < 25; ++v) { col[v] = __expf(col[v] - mx); s += col[v]; }
            const float inv = 1.f / s;
#pragma unroll
            for (int v = 0; v < 25; ++v) {
                float adj = A[(k * 25 + v) * 25 + u] + Bm[(k * 25 + v) * 25 + u] + col[v] * inv;
                adjbf[((k * 64 + b) * 32 + u) * 32 + v] = (__bf16)adj;
            }
            for (int v = 25; v < 32; ++v) adjbf[((k * 64 + b) * 32 + u) * 32 + v] = (__bf16)0.f;
        } else {
            for (int v = 0; v < 32; ++v) adjbf[((k * 64 + b) * 32 + u) * 32 + v] = (__bf16)0.f;
        }
    }
}

// ---------------------------------------------------------------------------
// Kernel 2: fused xa = x*adj and y = sum_k w_out*xa + bias, BN, ReLU.
// Block = (b, t-chunk of 8); 8 waves, each owns one 16-row o-tile of y.
// adj tiles are pulled into LDS with the Tensor Data Mover.
// ---------------------------------------------------------------------------
__global__ __launch_bounds__(256) void k_output(
    const float* __restrict__ x,       // (64,64,300,25)
    const __bf16* __restrict__ adjbf,  // (3,64,32,32)
    const __bf16* __restrict__ wo_bf,  // (3,128,64)
    const float* __restrict__ b_out,   // (3,128)
    const float* __restrict__ gamma, const float* __restrict__ beta,
    const float* __restrict__ mean,  const float* __restrict__ var,
    float* __restrict__ y)             // (64,128,300,25)
{
    __shared__ __bf16 xs[512 * 32];   // [m=(tl*64+c)][v pad32], 16B XOR-swizzled
    __shared__ __bf16 xa[208 * 72];   // [n=(tl*25+u) pad208][c], stride 144B (conflict-free)
    __shared__ __bf16 adjl[1024];     // current k adjacency tile (TDM target)

    const int b  = blockIdx.x / 38;
    const int t0 = (blockIdx.x % 38) * 8;
    const int tid  = threadIdx.x;
    const int wave = tid >> 5, lane = tid & 31;
    const int l15  = lane & 15, hh = lane >> 4;
    const v8f vzero = {0.f, 0.f, 0.f, 0.f, 0.f, 0.f, 0.f, 0.f};

    // zero pads (swizzled addresses must match the loads)
    for (int m = tid; m < 512; m += 256)
        for (int v = 25; v < 32; ++v)
            xs[m * 32 + (((v >> 3) ^ (m & 3)) << 3) + (v & 7)] = (__bf16)0.f;
    for (int i = tid; i < 8 * 72; i += 256) xa[200 * 72 + i] = (__bf16)0.f;

    // stage x chunk: [m=(tl*64+c)][v] bf16, swizzled
    for (int idx = tid; idx < 64 * 200; idx += 256) {
        int c = idx / 200, j = idx % 200;
        int tl = j / 25, v = j % 25;
        int m = tl * 64 + c;
        float val = (t0 * 25 + j < 7500) ? x[(b * 64 + c) * 7500 + t0 * 25 + j] : 0.f;
        xs[m * 32 + (((v >> 3) ^ (m & 3)) << 3) + (v & 7)] = (__bf16)val;
    }

    v8f yacc[13];
#pragma unroll
    for (int nt = 0; nt < 13; ++nt) yacc[nt] = vzero;

    for (int k = 0; k < 3; ++k) {
        // TDM-load this k's (b) adjacency tile (2KB) into LDS, then sync
        if (wave == 0) {
            tdm_load_2kb(adjbf + (k * 64 + b) * 1024, adjl);
            __builtin_amdgcn_s_wait_tensorcnt(0);
        }
        __syncthreads();   // also orders xs staging on k==0

        // hoisted adj B-fragments (reused by 4 m-tiles per wave)
        v16bf bfa[2];
#pragma unroll
        for (int nt2 = 0; nt2 < 2; ++nt2) {
            const __bf16* brow = adjl + (nt2 * 16 + l15) * 32;
            bfa[nt2] = frag_2x8(brow + 8 * hh, brow + 16 + 8 * hh);
        }

        // GEMM1: xa[(tl,u)][c] = xs(512x32) * adj(32x25pad)
#pragma unroll
        for (int mm = 0; mm < 4; ++mm) {
            const int mt = wave * 4 + mm;
            const int mrow = mt * 16 + l15;
            const __bf16* ab = xs + mrow * 32;
            const int sw = mrow & 3;
            v16bf af = frag_2x8(ab + ((hh ^ sw) << 3), ab + (((2 + hh) ^ sw) << 3));
#pragma unroll
            for (int nt2 = 0; nt2 < 2; ++nt2) {
                v8f d = wmma_bf16(af, bfa[nt2], vzero);
                const int u = nt2 * 16 + l15;
                if (u < 25) {
                    const int m0 = mt * 16 + 8 * hh;      // 8 consecutive m -> 8 consecutive c
                    const int c0 = m0 & 63, tl2 = m0 >> 6;
                    union { uint4 q; __bf16 e[8]; } pk;
#pragma unroll
                    for (int r = 0; r < 8; ++r) pk.e[r] = (__bf16)d[r];
                    *(uint4*)&xa[(tl2 * 25 + u) * 72 + c0] = pk.q;   // 16B packed store
                }
            }
        }
        __syncthreads();

        // GEMM2: y[o][(t,u)] += w_out(128x64) * xa(64x200); wave owns o-tile
        const __bf16* arow0 = wo_bf + (k * 128 + wave * 16 + l15) * 64;
#pragma unroll
        for (int kk = 0; kk < 2; ++kk) {
            v16bf af = frag_2x8(arow0 + kk * 32 + 8 * hh, arow0 + kk * 32 + 16 + 8 * hh);
#pragma unroll
            for (int nt = 0; nt < 13; ++nt) {
                const __bf16* brow = xa + (nt * 16 + l15) * 72 + kk * 32 + 8 * hh;
                yacc[nt] = wmma_bf16(af, frag_2x8(brow, brow + 16), yacc[nt]);
            }
        }
        __syncthreads();
    }

    // epilogue: bias + BN + ReLU (per-r factors precomputed once)
    float bv[8], sv[8], hv[8];
#pragma unroll
    for (int r = 0; r < 8; ++r) {
        const int o = wave * 16 + 8 * hh + r;
        bv[r] = b_out[o] + b_out[128 + o] + b_out[256 + o];
        sv[r] = gamma[o] * rsqrtf(var[o] + 1e-5f);
        hv[r] = beta[o] - mean[o] * sv[r];
    }
#pragma unroll
    for (int nt = 0; nt < 13; ++nt) {
        const int n = nt * 16 + l15;
        if (n < 200) {
            const int tl = n / 25, u = n % 25;
            const int t = t0 + tl;
            if (t < 300) {
                float* yp = y + ((b * 128 + wave * 16 + 8 * hh) * 300 + t) * 25 + u;
#pragma unroll
                for (int r = 0; r < 8; ++r)
                    yp[r * 7500] = fmaxf((yacc[nt][r] + bv[r]) * sv[r] + hv[r], 0.f);
            }
        }
    }
}

// ---------------------------------------------------------------------------
extern "C" void kernel_launch(void* const* d_in, const int* in_sizes, int n_in,
                              void* d_out, int out_size, void* d_ws, size_t ws_size,
                              hipStream_t stream) {
    const float* x       = (const float*)d_in[0];
    const float* A       = (const float*)d_in[1];
    const float* Bm      = (const float*)d_in[2];
    const float* w_theta = (const float*)d_in[3];
    const float* b_theta = (const float*)d_in[4];
    const float* w_phi   = (const float*)d_in[5];
    const float* b_phi   = (const float*)d_in[6];
    const float* w_out   = (const float*)d_in[7];
    const float* b_out   = (const float*)d_in[8];
    const float* bn_g    = (const float*)d_in[9];
    const float* bn_b    = (const float*)d_in[10];
    const float* bn_m    = (const float*)d_in[11];
    const float* bn_v    = (const float*)d_in[12];
    float* yout = (float*)d_out;

    __bf16* wsb    = (__bf16*)d_ws;
    __bf16* wth_bf = wsb;              // 6144 elems
    __bf16* wph_bf = wsb + 6144;       // 6144
    __bf16* wo_bf  = wsb + 12288;      // 24576
    __bf16* adjbf  = wsb + 36864;      // 3*64*32*32 = 196608

    k_convert_weights<<<144, 256, 0, stream>>>(w_theta, w_phi, w_out, wth_bf, wph_bf, wo_bf);
    k_attention<<<192, 256, 0, stream>>>(x, A, Bm, wth_bf, wph_bf, b_theta, b_phi, adjbf);
    k_output<<<64 * 38, 256, 0, stream>>>(x, adjbf, wo_bf, b_out, bn_g, bn_b, bn_m, bn_v, yout);
}
